// GNN_2946347565062
// MI455X (gfx1250) — compile-verified, matrix operand order checked by reading, not run on hardware
//
#include <hip/hip_runtime.h>
#include <hip/hip_bf16.h>
#include <cstdint>

#define FD 128  // feature dim

typedef __attribute__((ext_vector_type(16))) _Float16 v16h;
typedef __attribute__((ext_vector_type(8)))  float    v8f;

// ---- monotone unsigned key for float atomic-max ----
__device__ __forceinline__ unsigned fkey(float f) {
    unsigned u = __float_as_uint(f);
    return (u & 0x80000000u) ? ~u : (u | 0x80000000u);
}
__device__ __forceinline__ float funkey(unsigned k) {
    return (k & 0x80000000u) ? __uint_as_float(k ^ 0x80000000u)
                             : __uint_as_float(~k);
}

// ---- zero/seed accumulators ----
__global__ void k_init(float* agg, float* denom, unsigned* emax, float* normsq, int n) {
    int i = blockIdx.x * blockDim.x + threadIdx.x;
    if (i < n * FD) agg[i] = 0.f;
    if (i < n) { denom[i] = 0.f; emax[i] = 0u; }
    if (i == 0) *normsq = 0.f;
}

// ---- h = x @ W^T ; a_src = h.att_src ; a_dst = h.att_dst ----
__global__ void __launch_bounds__(FD) k_linear(
        const float* __restrict__ x, const float* __restrict__ W,
        const float* __restrict__ attس, const float* __restrict__ attd,
        float* __restrict__ h, float* __restrict__ a_src, float* __restrict__ a_dst) {
    __shared__ float xr[FD];
    __shared__ float red0[FD];
    __shared__ float red1[FD];
    const int i = blockIdx.x, t = threadIdx.x;
    xr[t] = x[(size_t)i * FD + t];
    __syncthreads();
    const float* wr = W + (size_t)t * FD;   // row t of W  ->  h[i,t] = x[i,:].W[t,:]
    float acc = 0.f;
#pragma unroll 8
    for (int j = 0; j < FD; ++j) acc += xr[j] * wr[j];
    h[(size_t)i * FD + t] = acc;
    red0[t] = acc * attس[t];
    red1[t] = acc * attd[t];
    __syncthreads();
    for (int s = FD / 2; s > 0; s >>= 1) {
        if (t < s) { red0[t] += red0[t + s]; red1[t] += red1[t + s]; }
        __syncthreads();
    }
    if (t == 0) { a_src[i] = red0[0]; a_dst[i] = red1[0]; }
}

// ---- pass 1: leaky-relu scores + per-dst segment max ----
__global__ void k_edge_max(const int* __restrict__ ei, int E, int n,
                           const float* __restrict__ a_src, const float* __restrict__ a_dst,
                           float* __restrict__ scores, unsigned* __restrict__ emax) {
    int e = blockIdx.x * blockDim.x + threadIdx.x;
    int ET = E + n;
    if (e >= ET) return;
    int s, d;
    if (e < E) { s = ei[e]; d = ei[E + e]; } else { s = d = e - E; }   // self loops appended
    float v = a_src[s] + a_dst[d];
    v = v > 0.f ? v : 0.2f * v;          // negative_slope = 0.2
    scores[e] = v;
    atomicMax(&emax[d], fkey(v));
}

// ---- pass 2: exp(score - max) + per-dst denom ----
__global__ void k_edge_exp(const int* __restrict__ ei, int E, int n,
                           const float* __restrict__ scores, const unsigned* __restrict__ emax,
                           float* __restrict__ eexp, float* __restrict__ denom) {
    int e = blockIdx.x * blockDim.x + threadIdx.x;
    int ET = E + n;
    if (e >= ET) return;
    int d = (e < E) ? ei[E + e] : (e - E);
    float ex = __expf(scores[e] - funkey(emax[d]));
    eexp[e] = ex;
    atomicAdd(&denom[d], ex);
}

// ---- pass 3: agg[d,:] += alpha * h[s,:]  (one block per edge, 128 lanes) ----
__global__ void __launch_bounds__(FD) k_edge_agg(
        const int* __restrict__ ei, int E, int n,
        const float* __restrict__ eexp, const float* __restrict__ denom,
        const float* __restrict__ h, float* __restrict__ agg) {
    int e = blockIdx.x, t = threadIdx.x;
    int s, d;
    if (e < E) { s = ei[e]; d = ei[E + e]; } else { s = d = e - E; }
    float alpha = eexp[e] / (denom[d] + 1e-16f);
    atomicAdd(&agg[(size_t)d * FD + t], alpha * h[(size_t)s * FD + t]);
}

// ---- +bias, leaky(0.02), +residual, sum of squares for Frobenius norm ----
__global__ void __launch_bounds__(256) k_finalize(
        const float* __restrict__ x, const float* __restrict__ bias,
        float* __restrict__ agg, float* __restrict__ normsq, int n) {
    __shared__ float red[256];
    int idx = blockIdx.x * 256 + threadIdx.x;
    float g = 0.f;
    if (idx < n * FD) {
        g = agg[idx] + bias[idx & (FD - 1)];
        g = g > 0.f ? g : 0.02f * g;
        g += x[idx];
        agg[idx] = g;
    }
    red[threadIdx.x] = g * g;
    __syncthreads();
    for (int s = 128; s > 0; s >>= 1) {
        if (threadIdx.x < s) red[threadIdx.x] += red[threadIdx.x + s];
        __syncthreads();
    }
    if (threadIdx.x == 0) atomicAdd(normsq, red[0]);
}

// ---- y = g / ||g||_F, cast to f16, zero-pad rows to npad ----
__global__ void k_cast(const float* __restrict__ agg, const float* __restrict__ normsq,
                       _Float16* __restrict__ yh, int n, int npad) {
    int idx = blockIdx.x * 256 + threadIdx.x;
    if (idx >= npad * FD) return;
    float inv = rsqrtf(*normsq);
    int row = idx >> 7;
    float v = (row < n) ? agg[idx] * inv : 0.f;
    yh[idx] = (_Float16)v;
}

// ---- pred = y @ y^T via v_wmma_f32_16x16x32_f16 ----
// 8 waves/block; each wave owns a 32x32 tile (2x2 of 16x16 accumulators);
// block tile = 64 rows x 128 cols. npad is a multiple of 128 so all fragment
// loads are in-bounds; output stores are guarded to n.
__global__ void __launch_bounds__(256) k_gemm(
        const _Float16* __restrict__ Y, float* __restrict__ P, int n, int npad) {
    union Frag { v16h v; uint4 q[2]; };
    const int wave = threadIdx.x >> 5;
    const int lane = threadIdx.x & 31;
    const int iBlk = wave >> 2;          // 0..1
    const int jBlk = wave & 3;           // 0..3
    const int row0 = blockIdx.y * 64 + iBlk * 32;
    const int col0 = blockIdx.x * 128 + jBlk * 32;
    const int m  = lane & 15;
    const int hi = lane >> 4;

    // ISA 16-bit A-layout: lane<16 -> M=lane, K chunks [0..7],[16..23];
    // lane>=16 -> M=lane-16, K chunks [8..15],[24..31].  B (=Y^T) uses the
    // same per-lane addressing on rows of Y (indexed by N instead of M).
    const _Float16* a0p = Y + (size_t)(row0 + m) * FD;
    const _Float16* a1p = Y + (size_t)(row0 + 16 + m) * FD;
    const _Float16* b0p = Y + (size_t)(col0 + m) * FD;
    const _Float16* b1p = Y + (size_t)(col0 + 16 + m) * FD;

    v8f c00 = {}, c01 = {}, c10 = {}, c11 = {};

#pragma unroll
    for (int k0 = 0; k0 < FD; k0 += 32) {
        const int o0 = k0 + hi * 8;
        const int o1 = k0 + 16 + hi * 8;
        Frag a0, a1, b0, b1;
        a0.q[0] = *(const uint4*)(a0p + o0); a0.q[1] = *(const uint4*)(a0p + o1);
        a1.q[0] = *(const uint4*)(a1p + o0); a1.q[1] = *(const uint4*)(a1p + o1);
        b0.q[0] = *(const uint4*)(b0p + o0); b0.q[1] = *(const uint4*)(b0p + o1);
        b1.q[0] = *(const uint4*)(b1p + o0); b1.q[1] = *(const uint4*)(b1p + o1);
        c00 = __builtin_amdgcn_wmma_f32_16x16x32_f16(false, a0.v, false, b0.v, (short)0, c00, false, false);
        c01 = __builtin_amdgcn_wmma_f32_16x16x32_f16(false, a0.v, false, b1.v, (short)0, c01, false, false);
        c10 = __builtin_amdgcn_wmma_f32_16x16x32_f16(false, a1.v, false, b0.v, (short)0, c10, false, false);
        c11 = __builtin_amdgcn_wmma_f32_16x16x32_f16(false, a1.v, false, b1.v, (short)0, c11, false, false);
    }

    // C layout: VGPR v, lanes 0-15: (M=v, N=lane); lanes 16-31: (M=v+8, N=lane-16)
#pragma unroll
    for (int ii = 0; ii < 2; ++ii) {
#pragma unroll
        for (int jj = 0; jj < 2; ++jj) {
            v8f c = ii ? (jj ? c11 : c10) : (jj ? c01 : c00);
            const int rbase = row0 + ii * 16 + hi * 8;
            const int cc = col0 + jj * 16 + m;
            if (cc >= n) continue;
#pragma unroll
            for (int v = 0; v < 8; ++v) {
                const int r = rbase + v;
                if (r < n) P[(size_t)r * n + cc] = c[v];
            }
        }
    }
}

extern "C" void kernel_launch(void* const* d_in, const int* in_sizes, int n_in,
                              void* d_out, int out_size, void* d_ws, size_t ws_size,
                              hipStream_t stream) {
    const float* x    = (const float*)d_in[0];
    const int*   ei   = (const int*)d_in[1];
    const float* W    = (const float*)d_in[2];
    const float* atts = (const float*)d_in[3];
    const float* attd = (const float*)d_in[4];
    const float* bias = (const float*)d_in[5];
    float* P = (float*)d_out;

    const int n  = in_sizes[0] / FD;      // 10000
    const int E  = in_sizes[1] / 2;       // 320000
    const int ET = E + n;                 // edges + self loops
    const int npad = ((n + 127) / 128) * 128;   // 10112

    // workspace layout (float units, 16-elem aligned => 64B)
    auto al = [](size_t v) { return (v + 15) & ~(size_t)15; };
    float* ws = (float*)d_ws;
    size_t offH    = 0;
    size_t offAs   = al(offH   + (size_t)n * FD);
    size_t offAd   = al(offAs  + n);
    size_t offEm   = al(offAd  + n);
    size_t offDen  = al(offEm  + n);
    size_t offSc   = al(offDen + n);
    size_t offEx   = al(offSc  + ET);
    size_t offAgg  = al(offEx  + ET);
    size_t offNorm = al(offAgg + (size_t)n * FD);
    size_t offY    = al(offNorm + 16);

    float*    h     = ws + offH;
    float*    a_src = ws + offAs;
    float*    a_dst = ws + offAd;
    unsigned* emax  = (unsigned*)(ws + offEm);
    float*    den   = ws + offDen;
    float*    sc    = ws + offSc;
    float*    ex    = ws + offEx;
    float*    agg   = ws + offAgg;
    float*    nrm   = ws + offNorm;
    _Float16* yh    = (_Float16*)(ws + offY);

    (void)n_in; (void)out_size; (void)ws_size;

    k_init<<<(n * FD + 255) / 256, 256, 0, stream>>>(agg, den, emax, nrm, n);
    k_linear<<<n, FD, 0, stream>>>(x, W, atts, attd, h, a_src, a_dst);
    k_edge_max<<<(ET + 255) / 256, 256, 0, stream>>>(ei, E, n, a_src, a_dst, sc, emax);
    k_edge_exp<<<(ET + 255) / 256, 256, 0, stream>>>(ei, E, n, sc, emax, ex, den);
    k_edge_agg<<<ET, FD, 0, stream>>>(ei, E, n, ex, den, h, agg);
    k_finalize<<<(n * FD + 255) / 256, 256, 0, stream>>>(x, bias, agg, nrm, n);
    k_cast<<<(npad * FD + 255) / 256, 256, 0, stream>>>(agg, nrm, yh, n, npad);

    dim3 grid(npad / 128, npad / 64);
    k_gemm<<<grid, 256, 0, stream>>>(yh, P, n, npad);
}
